// GraphAttentionEncoder_8065948582550
// MI455X (gfx1250) — matrix operand order repaired and checked by reading, not compiled, not run on hardware
//
#include <hip/hip_runtime.h>

typedef __bf16 bf16;
typedef __attribute__((ext_vector_type(16))) __bf16 v16bf;
typedef __attribute__((ext_vector_type(8)))  float  v8f;

union V16 {
  uint4  u[2];
  v16bf  v;
  __bf16 e[16];
};

__device__ __forceinline__ float warp_max16(float x) {
#pragma unroll
  for (int off = 1; off < 16; off <<= 1) x = fmaxf(x, __shfl_xor(x, off, 32));
  return x;
}
__device__ __forceinline__ float warp_sum16(float x) {
#pragma unroll
  for (int off = 1; off < 16; off <<= 1) x += __shfl_xor(x, off, 32);
  return x;
}

// ---------------------------------------------------------------------------
// f32 -> bf16 conversion
// ---------------------------------------------------------------------------
__global__ void f2bf(const float* __restrict__ s, bf16* __restrict__ d, long long n) {
  long long i = (long long)blockIdx.x * 256 + threadIdx.x;
  if (i < n) d[i] = (bf16)s[i];
}

// fill the end-of-token rows (row 512 of every batch item)
__global__ void eot_fill(const float* __restrict__ eot, float* h, bf16* hb) {
  int c = blockIdx.x * 256 + threadIdx.x;   // 0..511
  int b = blockIdx.y;
  long long off = ((long long)b * 513 + 512) * 512 + c;
  float v = eot[c];
  h[off]  = v;
  hb[off] = (bf16)v;
}

// ---------------------------------------------------------------------------
// Generic bf16 WMMA GEMM: C[M,N] = A[M,K] * B[K,N] (+bias)(+relu)
// 64x64 block tile, 4 waves, K step 32, LDS-staged A (row-major) and
// B (transposed so B fragments are contiguous per lane).
// Optional z-batching (strides sB, sC) and row padding (insert a gap row
// every `pad_every` rows -> used to write straight into the [b,513,512]
// concat layout from a [b*512] GEMM).
// ---------------------------------------------------------------------------
__global__ __launch_bounds__(128) void gemm_wmma(
    const bf16* __restrict__ A, const bf16* __restrict__ Bm,
    const float* __restrict__ bias,
    float* Cf, bf16* Cb,
    int M, int N, int K, int lda, int ldb, int ldc,
    long long sB, long long sC, int relu, int pad_every)
{
  __shared__ __align__(16) bf16 As[64 * 32];
  __shared__ __align__(16) bf16 Bs[64 * 32];   // transposed: Bs[n*32 + k]
  const bf16* Bp = Bm + (long long)blockIdx.z * sB;
  const int nb = blockIdx.x * 64;
  const int mb = blockIdx.y * 64;
  const int t = threadIdx.x;
  const int wave = t >> 5, lane = t & 31;
  const int lrow = lane & 15, half = lane >> 4;

  v8f acc[4] = {};
  for (int kb = 0; kb < K; kb += 32) {
    // stage A tile 64x32 (row-major)
#pragma unroll
    for (int i = 0; i < 2; ++i) {
      int idx = t + i * 128;
      int r   = idx >> 2;
      int cg  = (idx & 3) * 8;
      int gr  = mb + r;
      uint4 val = make_uint4(0u, 0u, 0u, 0u);
      if (gr < M) val = *reinterpret_cast<const uint4*>(A + (long long)gr * lda + kb + cg);
      *reinterpret_cast<uint4*>(&As[r * 32 + cg]) = val;
    }
    // stage B tile 32x64 transposed -> Bs[n][k]
#pragma unroll
    for (int i = 0; i < 2; ++i) {
      int idx = t + i * 128;
      int kr  = idx >> 3;
      int ng  = (idx & 7) * 8;
      uint4 val = *reinterpret_cast<const uint4*>(Bp + (long long)(kb + kr) * ldb + nb + ng);
      const bf16* e = reinterpret_cast<const bf16*>(&val);
#pragma unroll
      for (int j = 0; j < 8; ++j) Bs[(ng + j) * 32 + kr] = e[j];
    }
    // prefetch next K tile (speculative -> safe)
    if (kb + 32 < K) {
      int gr = mb + (t & 63); if (gr >= M) gr = M - 1;
      __builtin_prefetch(A + (long long)gr * lda + kb + 32, 0, 0);
      __builtin_prefetch(Bp + (long long)(kb + 32 + (t & 31)) * ldb + nb, 0, 0);
    }
    __syncthreads();

    // A fragment: lane row = wave*16 + lrow; e[0..7]=K half*8.., e[8..15]=K 16+half*8..
    V16 a;
    const int am = wave * 16 + lrow;
    a.u[0] = *reinterpret_cast<const uint4*>(&As[am * 32 + half * 8]);
    a.u[1] = *reinterpret_cast<const uint4*>(&As[am * 32 + 16 + half * 8]);
#pragma unroll
    for (int nt = 0; nt < 4; ++nt) {
      V16 bfr;
      const int c = nt * 16 + lrow;
      bfr.u[0] = *reinterpret_cast<const uint4*>(&Bs[c * 32 + half * 8]);
      bfr.u[1] = *reinterpret_cast<const uint4*>(&Bs[c * 32 + 16 + half * 8]);
      acc[nt] = __builtin_amdgcn_wmma_f32_16x16x32_bf16(
          false, a.v, false, bfr.v, (short)0, acc[nt], false, false);
    }
    __syncthreads();
  }

  float* Cfp = Cf ? Cf + (long long)blockIdx.z * sC : nullptr;
  bf16*  Cbp = Cb ? Cb + (long long)blockIdx.z * sC : nullptr;
#pragma unroll
  for (int nt = 0; nt < 4; ++nt) {
    const int c = nb + nt * 16 + lrow;
    if (c >= N) continue;
    const float bv = bias ? bias[c] : 0.f;
#pragma unroll
    for (int v = 0; v < 8; ++v) {
      const int r = mb + wave * 16 + v + 8 * half;
      if (r >= M) continue;
      long long cr = r;
      if (pad_every) cr += r / pad_every;
      float x = acc[nt][v] + bv;
      if (relu) x = fmaxf(x, 0.f);
      const long long off = cr * ldc + c;
      if (Cfp) Cfp[off] = x;
      if (Cbp) Cbp[off] = (bf16)x;
    }
  }
}

// ---------------------------------------------------------------------------
// Flash attention: one wave per 16-query tile, 32-key tiles, head_dim 64.
// Q/K/V: [H][B][513][64] bf16.  O: [B][513][H*64] bf16 (heads-major cols so
// the Wo projection becomes a single [R,512]x[512,512] GEMM).
// ---------------------------------------------------------------------------
__global__ __launch_bounds__(128) void attn_flash(
    const bf16* __restrict__ Q, const bf16* __restrict__ Km,
    const bf16* __restrict__ Vm, bf16* O)
{
  __shared__ __align__(16) bf16 Vs[4][32 * 64];
  __shared__ __align__(16) bf16 Ps[4][16 * 32];
  const int wave = threadIdx.x >> 5, lane = threadIdx.x & 31;
  const int lrow = lane & 15, half = lane >> 4;
  const int qt = blockIdx.x * 4 + wave;
  if (qt * 16 >= 513) return;                 // uniform per wave, no barriers used
  const int b = blockIdx.y, h = blockIdx.z;
  bf16* vsl = Vs[wave];
  bf16* psl = Ps[wave];
  const long long hb = (long long)h * 32 + b;
  const bf16* Qp = Q  + hb * 513 * 64;
  const bf16* Kp = Km + hb * 513 * 64;
  const bf16* Vp = Vm + hb * 513 * 64;

  int qr = qt * 16 + lrow; if (qr > 512) qr = 512;
  V16 aq0, aq1;                                // Q A-fragments, kdim 0..31 / 32..63
  aq0.u[0] = *(const uint4*)(Qp + (long long)qr * 64 + half * 16);
  aq0.u[1] = *(const uint4*)(Qp + (long long)qr * 64 + half * 16 + 8);
  aq1.u[0] = *(const uint4*)(Qp + (long long)qr * 64 + 32 + half * 16);
  aq1.u[1] = *(const uint4*)(Qp + (long long)qr * 64 + 32 + half * 16 + 8);

  float rmax[8], rsum[8];
#pragma unroll
  for (int v = 0; v < 8; ++v) { rmax[v] = -1e30f; rsum[v] = 0.f; }
  v8f o[4] = {};

  for (int kt = 0; kt < 17; ++kt) {
    const int kb0 = kt * 32;
    // stage V tile [32 keys][64 dims] row-major (zero padded rows)
    {
      int krow = kb0 + lane;
      int src = krow <= 512 ? krow : 512;
#pragma unroll
      for (int j = 0; j < 8; ++j) {
        uint4 val = *(const uint4*)(Vp + (long long)src * 64 + j * 8);
        if (krow > 512) val = make_uint4(0u, 0u, 0u, 0u);
        *(uint4*)(vsl + lane * 64 + j * 8) = val;
      }
    }
    // S = Q K^T for two 16-key groups (K=64 split into 2 WMMAs each)
    const int key0 = kb0 + lrow, key1 = kb0 + 16 + lrow;
    const int ks0 = key0 <= 512 ? key0 : 512;
    const int ks1 = key1 <= 512 ? key1 : 512;
    V16 bk;
    v8f z = {};
    bk.u[0] = *(const uint4*)(Kp + (long long)ks0 * 64 + half * 16);
    bk.u[1] = *(const uint4*)(Kp + (long long)ks0 * 64 + half * 16 + 8);
    v8f s0 = __builtin_amdgcn_wmma_f32_16x16x32_bf16(false, aq0.v, false, bk.v, (short)0, z, false, false);
    bk.u[0] = *(const uint4*)(Kp + (long long)ks0 * 64 + 32 + half * 16);
    bk.u[1] = *(const uint4*)(Kp + (long long)ks0 * 64 + 32 + half * 16 + 8);
    s0 = __builtin_amdgcn_wmma_f32_16x16x32_bf16(false, aq1.v, false, bk.v, (short)0, s0, false, false);
    bk.u[0] = *(const uint4*)(Kp + (long long)ks1 * 64 + half * 16);
    bk.u[1] = *(const uint4*)(Kp + (long long)ks1 * 64 + half * 16 + 8);
    v8f s1 = __builtin_amdgcn_wmma_f32_16x16x32_bf16(false, aq0.v, false, bk.v, (short)0, z, false, false);
    bk.u[0] = *(const uint4*)(Kp + (long long)ks1 * 64 + 32 + half * 16);
    bk.u[1] = *(const uint4*)(Kp + (long long)ks1 * 64 + 32 + half * 16 + 8);
    s1 = __builtin_amdgcn_wmma_f32_16x16x32_bf16(false, aq1.v, false, bk.v, (short)0, s1, false, false);

    const bool val0 = key0 <= 512, val1 = key1 <= 512;
#pragma unroll
    for (int v = 0; v < 8; ++v) {
      float a0 = val0 ? s0[v] * 0.125f : -1e30f;      // scale = 1/sqrt(64)
      float a1 = val1 ? s1[v] * 0.125f : -1e30f;
      float m  = warp_max16(fmaxf(a0, a1));
      float nm = fmaxf(rmax[v], m);
      float corr = __expf(rmax[v] - nm);
      float p0 = val0 ? __expf(a0 - nm) : 0.f;
      float p1 = val1 ? __expf(a1 - nm) : 0.f;
      float pr = warp_sum16(p0 + p1);
      rsum[v] = rsum[v] * corr + pr;
      rmax[v] = nm;
#pragma unroll
      for (int nt = 0; nt < 4; ++nt) o[nt][v] *= corr;
      int mr = v + 8 * half;
      psl[mr * 32 + lrow]      = (bf16)p0;            // C-layout -> LDS (transpose)
      psl[mr * 32 + 16 + lrow] = (bf16)p1;
    }
    asm volatile("s_wait_dscnt 0x0" ::: "memory");    // same-wave LDS RAW fence

    V16 pa;                                           // P as A-fragment
    pa.u[0] = *(const uint4*)(psl + lrow * 32 + half * 16);
    pa.u[1] = *(const uint4*)(psl + lrow * 32 + half * 16 + 8);
#pragma unroll
    for (int nt = 0; nt < 4; ++nt) {
      V16 bv;
      const int c = nt * 16 + lrow;
#pragma unroll
      for (int j = 0; j < 16; ++j) bv.e[j] = vsl[(half * 16 + j) * 64 + c];
      o[nt] = __builtin_amdgcn_wmma_f32_16x16x32_bf16(
          false, pa.v, false, bv.v, (short)0, o[nt], false, false);
    }
  }
  // normalize and write heads-major output
#pragma unroll
  for (int nt = 0; nt < 4; ++nt) {
    const int c = nt * 16 + lrow;
#pragma unroll
    for (int v = 0; v < 8; ++v) {
      int q = qt * 16 + v + 8 * half;
      if (q < 513) {
        float val = o[nt][v] / rsum[v];
        O[((long long)b * 513 + q) * 512 + h * 64 + c] = (bf16)val;
      }
    }
  }
}

// ---------------------------------------------------------------------------
// BatchNorm (train mode, batch stats) fused with the residual add.
// ---------------------------------------------------------------------------
__global__ void bn_zero(float* s) { s[blockIdx.x * 256 + threadIdx.x] = 0.f; }

__global__ __launch_bounds__(256) void bn_stats(const float* __restrict__ X,
                                                const float* __restrict__ Dl,
                                                float* stats, int rows) {
  const int c  = threadIdx.x;
  const int rb = blockIdx.x * 128;
  const int re = min(rb + 128, rows);
  float s0 = 0.f, s1 = 0.f, q0 = 0.f, q1 = 0.f;
  for (int r = rb; r < re; ++r) {
    long long base = (long long)r * 512;
    float a = X[base + c] + Dl[base + c];
    float b = X[base + c + 256] + Dl[base + c + 256];
    s0 += a; q0 += a * a; s1 += b; q1 += b * b;
  }
  atomicAdd(&stats[c], s0);
  atomicAdd(&stats[c + 256], s1);
  atomicAdd(&stats[512 + c], q0);
  atomicAdd(&stats[512 + c + 256], q1);
}

__global__ __launch_bounds__(256) void bn_apply(const float* X, const float* Dl,
                                                const float* __restrict__ stats,
                                                const float* __restrict__ g,
                                                const float* __restrict__ bta,
                                                float* Yf, bf16* Yb, int rows) {
  long long i = (long long)blockIdx.x * 256 + threadIdx.x;
  int c = (int)(i & 511);
  float inv  = 1.f / rows;
  float mean = stats[c] * inv;
  float var  = stats[512 + c] * inv - mean * mean;
  float rs   = rsqrtf(var + 1e-5f);
  float xv = X[i] + Dl[i];
  float y  = (xv - mean) * rs * g[c] + bta[c];
  Yf[i] = y;
  Yb[i] = (bf16)y;
}

// ---------------------------------------------------------------------------
// Orchestration
// ---------------------------------------------------------------------------
static inline char* carve(char*& p, size_t bytes) {
  char* r = p;
  p += (bytes + 255) & ~(size_t)255;
  return r;
}

extern "C" void kernel_launch(void* const* d_in, const int* in_sizes, int n_in,
                              void* d_out, int out_size, void* d_ws, size_t ws_size,
                              hipStream_t stream) {
  (void)in_sizes; (void)n_in; (void)out_size; (void)ws_size;
  const float* x     = (const float*)d_in[0];
  const float* Winit = (const float*)d_in[1];
  const float* binit = (const float*)d_in[2];
  const float* eot   = (const float*)d_in[3];
  const float* Wq    = (const float*)d_in[4];
  const float* Wk    = (const float*)d_in[5];
  const float* Wv    = (const float*)d_in[6];
  const float* Wo    = (const float*)d_in[7];
  const float* bn1g  = (const float*)d_in[8];
  const float* bn1b  = (const float*)d_in[9];
  const float* W1    = (const float*)d_in[10];
  const float* b1    = (const float*)d_in[11];
  const float* W2    = (const float*)d_in[12];
  const float* b2    = (const float*)d_in[13];
  const float* bn2g  = (const float*)d_in[14];
  const float* bn2b  = (const float*)d_in[15];
  float* out = (float*)d_out;

  constexpr int Bn = 32, SEQ = 513, D = 512, HD = 64, NH = 8, FFD = 2048, IND = 256, NL = 3;
  constexpr int R  = Bn * SEQ;      // 16416
  constexpr int M0 = Bn * 512;      // 16384

  char* p = (char*)d_ws;
  bf16*  xb  = (bf16*)carve(p, (size_t)M0 * IND * 2);
  bf16*  wib = (bf16*)carve(p, (size_t)IND * D * 2);
  bf16*  wqb = (bf16*)carve(p, (size_t)NL * NH * D * HD * 2);
  bf16*  wkb = (bf16*)carve(p, (size_t)NL * NH * D * HD * 2);
  bf16*  wvb = (bf16*)carve(p, (size_t)NL * NH * D * HD * 2);
  bf16*  wob = (bf16*)carve(p, (size_t)NL * NH * HD * D * 2);
  bf16*  w1b = (bf16*)carve(p, (size_t)NL * D * FFD * 2);
  bf16*  w2b = (bf16*)carve(p, (size_t)NL * FFD * D * 2);
  float* h   = (float*)carve(p, (size_t)R * D * 4);
  bf16*  hb  = (bf16*)carve(p, (size_t)R * D * 2);
  bf16*  qb  = (bf16*)carve(p, (size_t)NH * R * HD * 2);
  bf16*  kb  = (bf16*)carve(p, (size_t)NH * R * HD * 2);
  bf16*  vb  = (bf16*)carve(p, (size_t)NH * R * HD * 2);
  bf16*  ob  = (bf16*)carve(p, (size_t)R * D * 2);
  float* tmp = (float*)carve(p, (size_t)R * D * 4);
  bf16*  ff1 = (bf16*)carve(p, (size_t)R * FFD * 2);
  float* stats = (float*)carve(p, 1024 * 4);

  auto cvt = [&](const float* s, bf16* d, long long n) {
    f2bf<<<dim3((unsigned)((n + 255) / 256)), 256, 0, stream>>>(s, d, n);
  };
  cvt(x, xb, (long long)M0 * IND);
  cvt(Winit, wib, (long long)IND * D);
  cvt(Wq, wqb, (long long)NL * NH * D * HD);
  cvt(Wk, wkb, (long long)NL * NH * D * HD);
  cvt(Wv, wvb, (long long)NL * NH * D * HD);
  cvt(Wo, wob, (long long)NL * NH * HD * D);
  cvt(W1, w1b, (long long)NL * D * FFD);
  cvt(W2, w2b, (long long)NL * FFD * D);

  auto gemm = [&](const bf16* A, int lda, const bf16* Bw, int ldb, long long sB,
                  const float* bias, float* Cf, bf16* Cb, int ldc, long long sC,
                  int M, int N, int K, int relu, int pad, int batch) {
    dim3 grid(N / 64, (M + 63) / 64, batch);
    gemm_wmma<<<grid, 128, 0, stream>>>(A, Bw, bias, Cf, Cb, M, N, K,
                                        lda, ldb, ldc, sB, sC, relu, pad);
  };

  // init embed (writes straight into [b,513,512] layout via row padding) + eot rows
  gemm(xb, IND, wib, D, 0, binit, h, hb, D, 0, M0, D, IND, 0, 512, 1);
  eot_fill<<<dim3(2, Bn), 256, 0, stream>>>(eot, h, hb);

  for (int i = 0; i < NL; ++i) {
    const long long ws_ = (long long)D * HD;   // per-head weight stride
    gemm(hb, D, wqb + (long long)i * NH * ws_, HD, ws_, nullptr,
         nullptr, qb, HD, (long long)R * HD, R, HD, D, 0, 0, NH);
    gemm(hb, D, wkb + (long long)i * NH * ws_, HD, ws_, nullptr,
         nullptr, kb, HD, (long long)R * HD, R, HD, D, 0, 0, NH);
    gemm(hb, D, wvb + (long long)i * NH * ws_, HD, ws_, nullptr,
         nullptr, vb, HD, (long long)R * HD, R, HD, D, 0, 0, NH);

    attn_flash<<<dim3(9, Bn, NH), 128, 0, stream>>>(qb, kb, vb, ob);

    // output projection: [R,512] x [512,512] (Wo flattened [H][HD][D])
    gemm(ob, D, wob + (long long)i * D * D, D, 0, nullptr, tmp, nullptr, D, 0,
         R, D, D, 0, 0, 1);

    bn_zero<<<4, 256, 0, stream>>>(stats);
    bn_stats<<<(R + 127) / 128, 256, 0, stream>>>(h, tmp, stats, R);
    bn_apply<<<(R * D) / 256, 256, 0, stream>>>(h, tmp, stats,
                                                bn1g + i * D, bn1b + i * D, h, hb, R);

    gemm(hb, D, w1b + (long long)i * D * FFD, FFD, 0, b1 + i * FFD,
         nullptr, ff1, FFD, 0, R, FFD, D, 1, 0, 1);
    gemm(ff1, FFD, w2b + (long long)i * FFD * D, D, 0, b2 + i * D,
         tmp, nullptr, D, 0, R, D, FFD, 0, 0, 1);

    bn_zero<<<4, 256, 0, stream>>>(stats);
    bn_stats<<<(R + 127) / 128, 256, 0, stream>>>(h, tmp, stats, R);
    float* yout = (i == NL - 1) ? out : h;
    bn_apply<<<(R * D) / 256, 256, 0, stream>>>(h, tmp, stats,
                                                bn2g + i * D, bn2b + i * D, yout, hb, R);
  }
}
